// MixtralMoE_15899968930373
// MI455X (gfx1250) — compile-verified, matrix operand order checked by reading, not run on hardware
//
#include <hip/hip_runtime.h>
#include <stdint.h>

#define T_TOK 16384
#define HID   1024
#define FFN   3584
#define NEXP  8
#define NSLOT (T_TOK * 2)

typedef __bf16 bf16_t;
typedef bf16_t v16bf __attribute__((ext_vector_type(16)));
typedef bf16_t v8bf  __attribute__((ext_vector_type(8)));
typedef float  v8f   __attribute__((ext_vector_type(8)));
typedef float  v4f   __attribute__((ext_vector_type(4)));

// ---------------------------------------------------------------------------
// WMMA helpers
// ---------------------------------------------------------------------------
__device__ inline v8f wmma_bf16(v16bf a, v16bf b, v8f c) {
  return __builtin_amdgcn_wmma_f32_16x16x32_bf16(false, a, false, b,
                                                 (short)0, c, false, false);
}

// Load a 16x32 bf16 fragment (A or B operand) from an LDS tile with row
// stride 32 elements (64 bytes).  Per ISA layout: lanes 0-15 hold K
// {0..7,16..23} of row (lane&15); lanes 16-31 hold K {8..15,24..31}.
__device__ inline v16bf ld_frag(const bf16_t* lds, int row_base, int lane) {
  const int half = (lane >> 4) & 1;
  const int r    = row_base + (lane & 15);
  const bf16_t* p = lds + r * 32 + half * 8;
  v8bf lo = *(const v8bf*)(p);
  v8bf hi = *(const v8bf*)(p + 16);
  v16bf f;
#pragma unroll
  for (int i = 0; i < 8; ++i) { f[i] = lo[i]; f[i + 8] = hi[i]; }
  return f;
}

__device__ inline v8bf cvt8(v4f a, v4f b) {
  v8bf r;
#pragma unroll
  for (int c = 0; c < 4; ++c) { r[c] = (bf16_t)a[c]; r[c + 4] = (bf16_t)b[c]; }
  return r;
}

// ---------------------------------------------------------------------------
// Kernel 1: gating -- logits, softmax, top-2, per-expert counts
// ---------------------------------------------------------------------------
__global__ __launch_bounds__(256) void moe_gate(const float* __restrict__ hs,
                                                const float* __restrict__ gw,
                                                int* __restrict__ tope,
                                                float* __restrict__ topw,
                                                int* __restrict__ counts) {
  const int wave = threadIdx.x >> 5;
  const int lane = threadIdx.x & 31;
  const int t = blockIdx.x * 8 + wave;

  float acc[NEXP];
#pragma unroll
  for (int e = 0; e < NEXP; ++e) acc[e] = 0.f;

  const float* x = hs + (size_t)t * HID;
  for (int h = lane; h < HID; h += 32) {
    const float xv = x[h];
#pragma unroll
    for (int e = 0; e < NEXP; ++e) acc[e] += xv * gw[e * HID + h];
  }
#pragma unroll
  for (int e = 0; e < NEXP; ++e)
    for (int m = 16; m; m >>= 1) acc[e] += __shfl_xor(acc[e], m, 32);

  if (lane == 0) {
    float mx = acc[0];
#pragma unroll
    for (int e = 1; e < NEXP; ++e) mx = fmaxf(mx, acc[e]);
    float p[NEXP];
#pragma unroll
    for (int e = 0; e < NEXP; ++e) p[e] = __expf(acc[e] - mx);
    int e0 = 0;
#pragma unroll
    for (int e = 1; e < NEXP; ++e) if (p[e] > p[e0]) e0 = e;
    int e1 = (e0 == 0) ? 1 : 0;
#pragma unroll
    for (int e = 0; e < NEXP; ++e) if (e != e0 && p[e] > p[e1]) e1 = e;
    const float w0 = p[e0] / (p[e0] + p[e1]);
    tope[t * 2 + 0] = e0; topw[t * 2 + 0] = w0;
    tope[t * 2 + 1] = e1; topw[t * 2 + 1] = 1.f - w0;
    atomicAdd(&counts[e0], 1);
    atomicAdd(&counts[e1], 1);
  }
}

// ---------------------------------------------------------------------------
// Kernel 2: exclusive prefix sum of expert counts
// ---------------------------------------------------------------------------
__global__ void moe_scan(const int* __restrict__ counts,
                         int* __restrict__ bases, int* __restrict__ cursor) {
  int b = 0;
  for (int e = 0; e < NEXP; ++e) {
    bases[e] = b;
    cursor[e] = b;
    b += counts[e];
  }
  bases[NEXP] = b;
}

// ---------------------------------------------------------------------------
// Kernel 3: scatter slot ids into compacted per-expert row lists
// ---------------------------------------------------------------------------
__global__ __launch_bounds__(256) void moe_scatter(const int* __restrict__ tope,
                                                   int* __restrict__ cursor,
                                                   int* __restrict__ rows) {
  const int s = blockIdx.x * 256 + threadIdx.x;
  if (s < NSLOT) {
    const int e = tope[s];
    const int pos = atomicAdd(&cursor[e], 1);
    rows[pos] = s;
  }
}

// ---------------------------------------------------------------------------
// Kernel 4: A = silu(X W1^T) * (X W3^T)  (gathered, per expert, bf16 out)
// Block tile: M=256 gathered rows, N=64 FFN cols, K stepped by 32 over H.
// Each wave owns two 16-row M-tiles so every B fragment feeds 2 WMMAs.
// ---------------------------------------------------------------------------
__global__ __launch_bounds__(256) void moe_ffn1(
    const float* __restrict__ hs, const float* __restrict__ w1,
    const float* __restrict__ w3, const int* __restrict__ counts,
    const int* __restrict__ bases, const int* __restrict__ rows,
    bf16_t* __restrict__ Abuf) {
  const int e = blockIdx.z;
  const int ne = counts[e];
  const int m0 = blockIdx.x * 256;
  if (m0 >= ne) return;
  const int f0 = blockIdx.y * 64;
  const int base = bases[e];

  __shared__ __align__(16) bf16_t Xs[256 * 32];   // 16 KB
  __shared__ __align__(16) bf16_t W1s[64 * 32];   //  4 KB
  __shared__ __align__(16) bf16_t W3s[64 * 32];   //  4 KB

  const int tid = threadIdx.x;
  const int wave = tid >> 5, lane = tid & 31;

  // X staging: one full 32-float row per thread.
  const int xrow = tid;
  const bool xvalid = (m0 + xrow) < ne;
  const float* xptr = hs;  // dummy
  if (xvalid) {
    const int slot = rows[base + m0 + xrow];
    xptr = hs + (size_t)(slot >> 1) * HID;
  }
  // W staging: 4 threads per row, 8 floats each.
  const int wrow = tid >> 2, wq = tid & 3;
  const float* w1p = w1 + (size_t)e * FFN * HID + (size_t)(f0 + wrow) * HID + wq * 8;
  const float* w3p = w3 + (size_t)e * FFN * HID + (size_t)(f0 + wrow) * HID + wq * 8;

  v8f accG[2][4], accU[2][4];
#pragma unroll
  for (int m = 0; m < 2; ++m)
#pragma unroll
    for (int n = 0; n < 4; ++n)
#pragma unroll
      for (int j = 0; j < 8; ++j) { accG[m][n][j] = 0.f; accU[m][n][j] = 0.f; }

  for (int k0 = 0; k0 < HID; k0 += 32) {
    // --- stage X tile (gather + fp32->bf16), 32 cols per thread ---
    {
      v8bf q0, q1, q2, q3;
      if (xvalid) {
        const v4f* g = (const v4f*)(xptr + k0);
        q0 = cvt8(g[0], g[1]); q1 = cvt8(g[2], g[3]);
        q2 = cvt8(g[4], g[5]); q3 = cvt8(g[6], g[7]);
      } else {
#pragma unroll
        for (int c = 0; c < 8; ++c) {
          q0[c] = (bf16_t)0.f; q1[c] = (bf16_t)0.f;
          q2[c] = (bf16_t)0.f; q3[c] = (bf16_t)0.f;
        }
      }
      v8bf* dst = (v8bf*)&Xs[xrow * 32];
      dst[0] = q0; dst[1] = q1; dst[2] = q2; dst[3] = q3;
    }
    // --- stage W1/W3 tiles ---
    {
      const v4f* g1 = (const v4f*)(w1p + k0);
      const v4f* g3 = (const v4f*)(w3p + k0);
      *(v8bf*)&W1s[wrow * 32 + wq * 8] = cvt8(g1[0], g1[1]);
      *(v8bf*)&W3s[wrow * 32 + wq * 8] = cvt8(g3[0], g3[1]);
    }
    __syncthreads();

    const v16bf a0 = ld_frag(Xs, wave * 32, lane);
    const v16bf a1 = ld_frag(Xs, wave * 32 + 16, lane);
#pragma unroll
    for (int nt = 0; nt < 4; ++nt) {
      const v16bf b1 = ld_frag(W1s, nt * 16, lane);
      accG[0][nt] = wmma_bf16(a0, b1, accG[0][nt]);
      accG[1][nt] = wmma_bf16(a1, b1, accG[1][nt]);
      const v16bf b3 = ld_frag(W3s, nt * 16, lane);
      accU[0][nt] = wmma_bf16(a0, b3, accU[0][nt]);
      accU[1][nt] = wmma_bf16(a1, b3, accU[1][nt]);
    }
    __syncthreads();
  }

  // Epilogue: silu(g)*u -> bf16 intermediate, compacted row layout.
  const int half = lane >> 4, nl = lane & 15;
#pragma unroll
  for (int mt = 0; mt < 2; ++mt)
#pragma unroll
    for (int nt = 0; nt < 4; ++nt)
#pragma unroll
      for (int j = 0; j < 8; ++j) {
        const int m = j + half * 8;
        const int rr = m0 + wave * 32 + mt * 16 + m;
        if (rr < ne) {
          const float g = accG[mt][nt][j];
          const float val = (g / (1.f + __expf(-g))) * accU[mt][nt][j];
          Abuf[(size_t)(base + rr) * FFN + f0 + nt * 16 + nl] = (bf16_t)val;
        }
      }
}

// ---------------------------------------------------------------------------
// Kernel 5: OUT += (A W2^T) * routing_weight, scatter-add to tokens.
// Double-buffered LDS: A tiles via async global->LDS, W2 via register
// pipeline; one barrier per K-step, async overlapped with WMMA.
// ---------------------------------------------------------------------------
__global__ __launch_bounds__(256) void moe_ffn2(
    const bf16_t* __restrict__ Abuf, const float* __restrict__ w2,
    const int* __restrict__ counts, const int* __restrict__ bases,
    const int* __restrict__ rows, const float* __restrict__ topw,
    float* __restrict__ out) {
  const int e = blockIdx.z;
  const int ne = counts[e];
  const int m0 = blockIdx.x * 256;
  if (m0 >= ne) return;
  const int h0 = blockIdx.y * 64;
  const int base = bases[e];

  __shared__ __align__(16) bf16_t Asm[2 * 256 * 32];  // 2 x 16 KB
  __shared__ __align__(16) bf16_t W2s[2 * 64 * 32];   // 2 x  4 KB

  const int tid = threadIdx.x;
  const int wave = tid >> 5, lane = tid & 31;
  const int wrow = tid >> 2, wq = tid & 3;
  const float* w2p = w2 + (size_t)e * HID * FFN + (size_t)(h0 + wrow) * FFN + wq * 8;

  // Precompute the 4 global A row addresses this thread streams (16B chunks).
  uint64_t ga[4];
  uint32_t la[4];
#pragma unroll
  for (int it = 0; it < 4; ++it) {
    const int c = tid + it * 256;  // 1024 chunks of 16B per 256x64B tile
    const int row = c >> 2;
    const int part = c & 3;
    int rr = m0 + row;
    if (rr > ne - 1) rr = ne - 1;  // clamp to stay in-bounds
    ga[it] = (uint64_t)(uintptr_t)(Abuf + (size_t)(base + rr) * FFN + part * 8);
    la[it] = (uint32_t)(uintptr_t)&Asm[row * 32 + part * 8];
  }

  v8f acc[2][4];
#pragma unroll
  for (int m = 0; m < 2; ++m)
#pragma unroll
    for (int n = 0; n < 4; ++n)
#pragma unroll
      for (int j = 0; j < 8; ++j) acc[m][n][j] = 0.f;

  // --- prologue: fill buffer 0 for k0 = 0 ---
#pragma unroll
  for (int it = 0; it < 4; ++it) {
    asm volatile("global_load_async_to_lds_b128 %0, %1, off"
                 :: "v"(la[it]), "v"(ga[it]) : "memory");
  }
  {
    const v4f* g = (const v4f*)(w2p);
    *(v8bf*)&W2s[wrow * 32 + wq * 8] = cvt8(g[0], g[1]);
  }
  asm volatile("s_wait_asynccnt 0x0" ::: "memory");
  __syncthreads();

  int cur = 0;
  for (int k0 = 0; k0 < FFN; k0 += 32) {
    const int nxt = k0 + 32;
    const bool more = nxt < FFN;
    v4f wreg0, wreg1;
    if (more) {
      // Issue next A tile (async -> other LDS buffer) and next W2 loads.
      const uint32_t lofs = (uint32_t)(cur ^ 1) * (256 * 32 * 2);
#pragma unroll
      for (int it = 0; it < 4; ++it) {
        asm volatile("global_load_async_to_lds_b128 %0, %1, off"
                     :: "v"(la[it] + lofs), "v"(ga[it] + (uint64_t)nxt * 2)
                     : "memory");
      }
      const v4f* g = (const v4f*)(w2p + nxt);
      wreg0 = g[0]; wreg1 = g[1];
    }
    // Compute on current buffer while the async engine streams the next one.
    const bf16_t* As = &Asm[cur * 256 * 32];
    const bf16_t* Ws = &W2s[cur * 64 * 32];
    const v16bf a0 = ld_frag(As, wave * 32, lane);
    const v16bf a1 = ld_frag(As, wave * 32 + 16, lane);
#pragma unroll
    for (int nt = 0; nt < 4; ++nt) {
      const v16bf b = ld_frag(Ws, nt * 16, lane);
      acc[0][nt] = wmma_bf16(a0, b, acc[0][nt]);
      acc[1][nt] = wmma_bf16(a1, b, acc[1][nt]);
    }
    if (more) {
      *(v8bf*)&W2s[(cur ^ 1) * 64 * 32 + wrow * 32 + wq * 8] = cvt8(wreg0, wreg1);
    }
    asm volatile("s_wait_asynccnt 0x0" ::: "memory");
    __syncthreads();
    cur ^= 1;
  }

  // Epilogue: scale by routing weight, scatter-add into token rows.
  const int half = lane >> 4, nl = lane & 15;
#pragma unroll
  for (int mt = 0; mt < 2; ++mt)
#pragma unroll
    for (int nt = 0; nt < 4; ++nt)
#pragma unroll
      for (int j = 0; j < 8; ++j) {
        const int m = j + half * 8;
        const int rr = m0 + wave * 32 + mt * 16 + m;
        if (rr < ne) {
          const int slot = rows[base + rr];
          const float wgt = topw[slot];
          const int t = slot >> 1;
          unsafeAtomicAdd(&out[(size_t)t * HID + h0 + nt * 16 + nl],
                          acc[mt][nt][j] * wgt);
        }
      }
}

// ---------------------------------------------------------------------------
// Launcher
// ---------------------------------------------------------------------------
extern "C" void kernel_launch(void* const* d_in, const int* in_sizes, int n_in,
                              void* d_out, int out_size, void* d_ws, size_t ws_size,
                              hipStream_t stream) {
  const float* hs = (const float*)d_in[0];
  const float* gw = (const float*)d_in[1];
  const float* w1 = (const float*)d_in[2];
  const float* w2 = (const float*)d_in[3];
  const float* w3 = (const float*)d_in[4];
  float* out = (float*)d_out;

  char* ws = (char*)d_ws;
  int*    counts = (int*)(ws + 0);
  int*    bases  = (int*)(ws + 64);
  int*    cursor = (int*)(ws + 128);
  int*    tope   = (int*)(ws + 256);
  float*  topw   = (float*)(ws + 256 + (size_t)NSLOT * 4);
  int*    rows   = (int*)(ws + 256 + (size_t)NSLOT * 8);
  bf16_t* Abuf   = (bf16_t*)(ws + 524288);  // 32768 x 3584 bf16 = 224 MB

  hipMemsetAsync(d_out, 0, (size_t)T_TOK * HID * sizeof(float), stream);
  hipMemsetAsync(ws, 0, 256, stream);

  moe_gate<<<T_TOK / 8, 256, 0, stream>>>(hs, gw, tope, topw, counts);
  moe_scan<<<1, 1, 0, stream>>>(counts, bases, cursor);
  moe_scatter<<<NSLOT / 256, 256, 0, stream>>>(tope, cursor, rows);
  moe_ffn1<<<dim3(T_TOK / 256, FFN / 64, NEXP), 256, 0, stream>>>(
      hs, w1, w3, counts, bases, rows, Abuf);
  moe_ffn2<<<dim3(T_TOK / 256, HID / 64, NEXP), 256, 0, stream>>>(
      Abuf, w2, counts, bases, rows, topw, out);
}
